// EdgeConvLayers_10196252361247
// MI455X (gfx1250) — matrix operand order repaired and checked by reading, not compiled, not run on hardware
//
#include <hip/hip_runtime.h>
#include <hip/hip_bf16.h>

// ---------------------------------------------------------------------------
// EdgeConv GNN on MI455X (gfx1250): bf16 WMMA GEMMs + f32 LN/ReLU epilogues.
// One wave32 per 16-row tile -> 16x128 output via 8x v_wmma_f32_16x16x32_bf16
// accumulators. Scatter-max aggregation via global u32 atomic max on float
// bits (valid: post-ReLU values >= 0, agg initialized to 0).
// f32->bf16 uses native v_cvt_pk_bf16_f32. Concat-segment selection is done
// on the (compile-time) k-tile index so no exec-mask branches are emitted.
// ---------------------------------------------------------------------------

#define NNODES 20000
#define NEDGES 320000
#define CH     128
#define LNEPS  1e-5f
#define LDPAD  136   // LDS row stride in bf16 elements (272B -> bank-skewed)

typedef __attribute__((ext_vector_type(16))) __bf16 v16bf;
typedef __attribute__((ext_vector_type(2)))  __bf16 v2bf;
typedef __attribute__((ext_vector_type(8)))  float  v8f;

union FragU { unsigned u[8]; v16bf v; };

__device__ inline unsigned short bf16bits(float f) {
  union { __bf16 h; unsigned short s; } x;
  x.h = (__bf16)f;           // hardware fptrunc, RNE
  return x.s;
}
__device__ inline unsigned pk_bf16(float a, float b) {
  union { v2bf h; unsigned u; } x;
  x.h[0] = (__bf16)a;        // fuses into v_cvt_pk_bf16_f32
  x.h[1] = (__bf16)b;
  return x.u;
}

// 8 consecutive columns -> 4 packed bf16 dwords
__device__ inline void chunk_direct(const float* row, int c, unsigned* o) {
  const float4* p = (const float4*)(row + c);
  float4 a = p[0], b = p[1];
  o[0] = pk_bf16(a.x, a.y); o[1] = pk_bf16(a.z, a.w);
  o[2] = pk_bf16(b.x, b.y); o[3] = pk_bf16(b.z, b.w);
}
__device__ inline void chunk_diff(const float* rj, const float* ri, int c, unsigned* o) {
  const float4* pj = (const float4*)(rj + c);
  const float4* pi = (const float4*)(ri + c);
  float4 a = pj[0], b = pj[1], x = pi[0], y = pi[1];
  o[0] = pk_bf16(a.x - x.x, a.y - x.y); o[1] = pk_bf16(a.z - x.z, a.w - x.w);
  o[2] = pk_bf16(b.x - y.x, b.y - y.y); o[3] = pk_bf16(b.z - y.z, b.w - y.w);
}

// msg = [x_i | x_j - x_i | e] (384 cols). t is a literal constant after
// unrolling, so the segment choice folds at compile time; only the address
// offset (sel*8) is per-lane.
__device__ inline v16bf afrag_msg(const float* xi, const float* xj, const float* ee,
                                  int t, int sel) {
  FragU f;
  int o0 = (t & 3) * 32 + sel * 8;   // offset inside the 128-col segment
  if (t < 4)      { chunk_direct(xi, o0, f.u);      chunk_direct(xi, o0 + 16, f.u + 4); }
  else if (t < 8) { chunk_diff(xj, xi, o0, f.u);    chunk_diff(xj, xi, o0 + 16, f.u + 4); }
  else            { chunk_direct(ee, o0, f.u);      chunk_direct(ee, o0 + 16, f.u + 4); }
  return f.v;
}
// et = [x_j - x_i | e] (256 cols)
__device__ inline v16bf afrag_et(const float* xi, const float* xj, const float* ee,
                                 int t, int sel) {
  FragU f;
  int o0 = (t & 3) * 32 + sel * 8;
  if (t < 4) { chunk_diff(xj, xi, o0, f.u);  chunk_diff(xj, xi, o0 + 16, f.u + 4); }
  else       { chunk_direct(ee, o0, f.u);    chunk_direct(ee, o0 + 16, f.u + 4); }
  return f.v;
}
// Direct 128-wide row (node head)
__device__ inline v16bf afrag_row(const float* row, int t, int sel) {
  FragU f; int c0 = t * 32 + sel * 8;
  chunk_direct(row, c0,      f.u);
  chunk_direct(row, c0 + 16, f.u + 4);
  return f.v;
}

// B fragment from fragment-order packed weights: ((t*8+j)*32+lane)*32 bytes.
__device__ inline v16bf bfrag(const unsigned* __restrict__ wp, int t, int j, int lane) {
  FragU f;
  const uint4* p = (const uint4*)(wp + ((size_t)((t * 8 + j) * 32 + lane)) * 8);
  uint4 a = p[0], b = p[1];
  f.u[0] = a.x; f.u[1] = a.y; f.u[2] = a.z; f.u[3] = a.w;
  f.u[4] = b.x; f.u[5] = b.y; f.u[6] = b.z; f.u[7] = b.w;
  return f.v;
}

// A fragment from staged LDS tile (16 x 128 bf16, stride LDPAD)
__device__ inline v16bf afrag_lds(const unsigned short* sm, int t, int lane) {
  int r = lane & 15, sel = lane >> 4;
  int c0 = t * 32 + sel * 8;
  FragU f;
  uint4 a = *(const uint4*)(sm + r * LDPAD + c0);
  uint4 b = *(const uint4*)(sm + r * LDPAD + c0 + 16);
  f.u[0] = a.x; f.u[1] = a.y; f.u[2] = a.z; f.u[3] = a.w;
  f.u[4] = b.x; f.u[5] = b.y; f.u[6] = b.z; f.u[7] = b.w;
  return f.v;
}

__device__ inline float hredux(float s) {  // reduce across 16-lane half
  s += __shfl_xor(s, 1, 32);
  s += __shfl_xor(s, 2, 32);
  s += __shfl_xor(s, 4, 32);
  s += __shfl_xor(s, 8, 32);
  return s;
}

// bias + LayerNorm + ReLU over 16x128 tile held in C-layout accumulators
__device__ inline void ln_relu(v8f* acc, const float* __restrict__ b,
                               const float* __restrict__ g,
                               const float* __restrict__ z, int lane) {
  int n = lane & 15;
  float bb[8], gg[8], be[8];
#pragma unroll
  for (int j = 0; j < 8; j++) { bb[j] = b[j*16+n]; gg[j] = g[j*16+n]; be[j] = z[j*16+n]; }
#pragma unroll
  for (int j = 0; j < 8; j++)
#pragma unroll
    for (int v = 0; v < 8; v++) acc[j][v] += bb[j];
#pragma unroll
  for (int v = 0; v < 8; v++) {
    float s = 0.f;
#pragma unroll
    for (int j = 0; j < 8; j++) s += acc[j][v];
    float mu = hredux(s) * (1.f / 128.f);
    float q = 0.f;
#pragma unroll
    for (int j = 0; j < 8; j++) { float d = acc[j][v] - mu; q += d * d; }
    float var = hredux(q) * (1.f / 128.f);
    float rs = 1.f / sqrtf(var + LNEPS);
#pragma unroll
    for (int j = 0; j < 8; j++) {
      float val = (acc[j][v] - mu) * rs * gg[j] + be[j];
      acc[j][v] = fmaxf(val, 0.f);
    }
  }
}

__device__ inline void stage_lds(unsigned short* sm, const v8f* acc, int lane) {
  int sel = lane >> 4, n = lane & 15;
#pragma unroll
  for (int v = 0; v < 8; v++) {
    int row = v + 8 * sel;
#pragma unroll
    for (int j = 0; j < 8; j++)
      sm[row * LDPAD + j * 16 + n] = bf16bits(acc[j][v]);
  }
}

#define WMMA_BF16(A, B, Cacc) \
  __builtin_amdgcn_wmma_f32_16x16x32_bf16(false, (A), false, (B), (short)0, (Cacc), false, false)

// ---------------------------------------------------------------------------
// Kernels
// ---------------------------------------------------------------------------

__global__ void copyf(float* __restrict__ d, const float* __restrict__ s, int nn) {
  int i = blockIdx.x * blockDim.x + threadIdx.x;
  if (i < nn) d[i] = s[i];
}
__global__ void zerof(float* __restrict__ d, int nn) {
  int i = blockIdx.x * blockDim.x + threadIdx.x;
  if (i < nn) d[i] = 0.f;
}
__global__ void addf(float* __restrict__ d, const float* __restrict__ s, int nn) {
  int i = blockIdx.x * blockDim.x + threadIdx.x;
  if (i < nn) d[i] += s[i];
}

// Pack W (K x 128, row-major f32) into fragment-order bf16.
__global__ void pack_w(const float* __restrict__ W, unsigned* __restrict__ dst, int KT) {
  int tid = blockIdx.x * blockDim.x + threadIdx.x;
  if (tid >= KT * 256) return;
  int lane = tid & 31, j = (tid >> 5) & 7, t = tid >> 8;
  int n = j * 16 + (lane & 15);
  int kbase = t * 32 + (lane >> 4) * 8;
  unsigned* o = dst + (size_t)tid * 8;
#pragma unroll
  for (int i = 0; i < 8; i++) {
    float w0 = W[(size_t)(kbase + 2 * (i & 3) + (i >> 2) * 16)     * CH + n];
    float w1 = W[(size_t)(kbase + 2 * (i & 3) + (i >> 2) * 16 + 1) * CH + n];
    o[i] = pk_bf16(w0, w1);
  }
}

// One conv layer: per 16-edge tile — node MLP (384->128->128) + scatter-max,
// edge MLP (256->128->128) + residual e update (row-in-place).
__global__ __launch_bounds__(32)
void conv_edge_kernel(const float* __restrict__ x, float* __restrict__ e,
                      const int* __restrict__ eidx,
                      const unsigned* __restrict__ wpN1, const float* bN1, const float* gN1, const float* zN1,
                      const unsigned* __restrict__ wpN2, const float* bN2, const float* gN2, const float* zN2,
                      const unsigned* __restrict__ wpE1, const float* bE1, const float* gE1, const float* zE1,
                      const unsigned* __restrict__ wpE2, const float* bE2, const float* gE2, const float* zE2,
                      float* __restrict__ agg) {
  __shared__ unsigned short sm[16 * LDPAD];
  const int lane = threadIdx.x;
  const int r = lane & 15, sel = lane >> 4, n = lane & 15;
  const int ebase = blockIdx.x * 16;
  const int me = ebase + r;
  const int srcN = eidx[me];
  const int tarN = eidx[NEDGES + me];
  const float* xj = x + (size_t)srcN * CH;
  const float* xi = x + (size_t)tarN * CH;
  const float* ee = e + (size_t)me * CH;
  const v8f vzero = {};
  v8f acc[8];

  // ---- node MLP layer 1: [E,384] @ [384,128] ----
#pragma unroll
  for (int j = 0; j < 8; j++) acc[j] = vzero;
#pragma unroll
  for (int t = 0; t < 12; t++) {
    v16bf A = afrag_msg(xi, xj, ee, t, sel);
#pragma unroll
    for (int j = 0; j < 8; j++) acc[j] = WMMA_BF16(A, bfrag(wpN1, t, j, lane), acc[j]);
  }
  ln_relu(acc, bN1, gN1, zN1, lane);
  stage_lds(sm, acc, lane);
  __syncthreads();

  // ---- node MLP layer 2: 128 -> 128 ----
#pragma unroll
  for (int j = 0; j < 8; j++) acc[j] = vzero;
#pragma unroll
  for (int t = 0; t < 4; t++) {
    v16bf A = afrag_lds(sm, t, lane);
#pragma unroll
    for (int j = 0; j < 8; j++) acc[j] = WMMA_BF16(A, bfrag(wpN2, t, j, lane), acc[j]);
  }
  ln_relu(acc, bN2, gN2, zN2, lane);

  // ---- scatter-max into agg (uint compare == float compare for >=0) ----
#pragma unroll
  for (int v = 0; v < 8; v++) {
    int row = v + 8 * sel;
    int tn = eidx[NEDGES + ebase + row];
    unsigned* arow = (unsigned*)(agg + (size_t)tn * CH);
#pragma unroll
    for (int j = 0; j < 8; j++)
      atomicMax(arow + j * 16 + n, __float_as_uint(acc[j][v]));
  }
  __syncthreads();

  // ---- edge MLP layer 1: [E,256] @ [256,128] ----
#pragma unroll
  for (int j = 0; j < 8; j++) acc[j] = vzero;
#pragma unroll
  for (int t = 0; t < 8; t++) {
    v16bf A = afrag_et(xi, xj, ee, t, sel);
#pragma unroll
    for (int j = 0; j < 8; j++) acc[j] = WMMA_BF16(A, bfrag(wpE1, t, j, lane), acc[j]);
  }
  ln_relu(acc, bE1, gE1, zE1, lane);
  stage_lds(sm, acc, lane);
  __syncthreads();

  // ---- edge MLP layer 2 ----
#pragma unroll
  for (int j = 0; j < 8; j++) acc[j] = vzero;
#pragma unroll
  for (int t = 0; t < 4; t++) {
    v16bf A = afrag_lds(sm, t, lane);
#pragma unroll
    for (int j = 0; j < 8; j++) acc[j] = WMMA_BF16(A, bfrag(wpE2, t, j, lane), acc[j]);
  }
  ln_relu(acc, bE2, gE2, zE2, lane);

  // ---- residual edge update (in place; each row owned by this wave) ----
#pragma unroll
  for (int v = 0; v < 8; v++) {
    int row = v + 8 * sel;
    float* erow = e + (size_t)(ebase + row) * CH;
#pragma unroll
    for (int j = 0; j < 8; j++) {
      int cc = j * 16 + n;
      erow[cc] = erow[cc] + acc[j][v];
    }
  }
}

// node head: block(x) (128->128, LN, ReLU) then Linear(128->128)+bias
__global__ __launch_bounds__(32)
void node_head_kernel(const float* __restrict__ x,
                      const unsigned* __restrict__ wpB, const float* bB, const float* gB, const float* zB,
                      const unsigned* __restrict__ wpO, const float* __restrict__ bO,
                      float* __restrict__ out) {
  __shared__ unsigned short sm[16 * LDPAD];
  const int lane = threadIdx.x;
  const int r = lane & 15, sel = lane >> 4, n = lane & 15;
  const int nbase = blockIdx.x * 16;
  const float* xr = x + (size_t)(nbase + r) * CH;
  const v8f vzero = {};
  v8f acc[8];
#pragma unroll
  for (int j = 0; j < 8; j++) acc[j] = vzero;
#pragma unroll
  for (int t = 0; t < 4; t++) {
    v16bf A = afrag_row(xr, t, sel);
#pragma unroll
    for (int j = 0; j < 8; j++) acc[j] = WMMA_BF16(A, bfrag(wpB, t, j, lane), acc[j]);
  }
  ln_relu(acc, bB, gB, zB, lane);
  stage_lds(sm, acc, lane);
  __syncthreads();
#pragma unroll
  for (int j = 0; j < 8; j++) acc[j] = vzero;
#pragma unroll
  for (int t = 0; t < 4; t++) {
    v16bf A = afrag_lds(sm, t, lane);
#pragma unroll
    for (int j = 0; j < 8; j++) acc[j] = WMMA_BF16(A, bfrag(wpO, t, j, lane), acc[j]);
  }
#pragma unroll
  for (int v = 0; v < 8; v++) {
    int row = v + 8 * sel;
    float* orow = out + (size_t)(nbase + row) * CH;
#pragma unroll
    for (int j = 0; j < 8; j++) {
      int cc = j * 16 + n;
      orow[cc] = acc[j][v] + bO[cc];
    }
  }
}

// edge head: block([xj-xi|e]) (256->128) then Linear(128->128)+bias, written
// in place over the e region of d_out.
__global__ __launch_bounds__(32)
void edge_head_kernel(const float* __restrict__ x, float* __restrict__ e,
                      const int* __restrict__ eidx,
                      const unsigned* __restrict__ wpB, const float* bB, const float* gB, const float* zB,
                      const unsigned* __restrict__ wpO, const float* __restrict__ bO) {
  __shared__ unsigned short sm[16 * LDPAD];
  const int lane = threadIdx.x;
  const int r = lane & 15, sel = lane >> 4, n = lane & 15;
  const int ebase = blockIdx.x * 16;
  const int me = ebase + r;
  const int srcN = eidx[me];
  const int tarN = eidx[NEDGES + me];
  const float* xj = x + (size_t)srcN * CH;
  const float* xi = x + (size_t)tarN * CH;
  const float* ee = e + (size_t)me * CH;
  const v8f vzero = {};
  v8f acc[8];
#pragma unroll
  for (int j = 0; j < 8; j++) acc[j] = vzero;
#pragma unroll
  for (int t = 0; t < 8; t++) {
    v16bf A = afrag_et(xi, xj, ee, t, sel);
#pragma unroll
    for (int j = 0; j < 8; j++) acc[j] = WMMA_BF16(A, bfrag(wpB, t, j, lane), acc[j]);
  }
  ln_relu(acc, bB, gB, zB, lane);
  stage_lds(sm, acc, lane);
  __syncthreads();
#pragma unroll
  for (int j = 0; j < 8; j++) acc[j] = vzero;
#pragma unroll
  for (int t = 0; t < 4; t++) {
    v16bf A = afrag_lds(sm, t, lane);
#pragma unroll
    for (int j = 0; j < 8; j++) acc[j] = WMMA_BF16(A, bfrag(wpO, t, j, lane), acc[j]);
  }
#pragma unroll
  for (int v = 0; v < 8; v++) {
    int row = v + 8 * sel;
    float* orow = e + (size_t)(ebase + row) * CH;
#pragma unroll
    for (int j = 0; j < 8; j++) {
      int cc = j * 16 + n;
      orow[cc] = acc[j][v] + bO[cc];
    }
  }
}

// ---------------------------------------------------------------------------
// Host launcher
// ---------------------------------------------------------------------------
extern "C" void kernel_launch(void* const* d_in, const int* in_sizes, int n_in,
                              void* d_out, int out_size, void* d_ws, size_t ws_size,
                              hipStream_t stream) {
  (void)in_sizes; (void)n_in; (void)out_size; (void)ws_size;
  const float* node_features = (const float*)d_in[0];
  const int*   eidx          = (const int*)d_in[1];
  const float* edge_features = (const float*)d_in[2];
  auto F = [&](int i) { return (const float*)d_in[i]; };

  // Param flat order (jax pytree: dicts by sorted key, lists/tuples in order):
  // convs[c]: edge_mlp0(W,b,g,beta)=pb+0..3, edge_mlp1=pb+4..7,
  //           node_mlp0=pb+8..11, node_mlp1=pb+12..15, pb=3+16c
  // edge_out: blk W,b,g,beta = 35..38; We,be = 39,40
  // node_out: blk = 41..44; Wn,bn = 45,46

  // Workspace layout
  float* x_cur = (float*)d_ws;                         // 20000*128
  float* agg   = x_cur + (size_t)NNODES * CH;          // 20000*128
  unsigned* wpool = (unsigned*)(agg + (size_t)NNODES * CH);
  size_t woff = 0;
  auto walloc = [&](int KT) { unsigned* p = wpool + woff; woff += (size_t)KT * 2048; return p; };

  // e lives (and finishes as edge_output) in the edge region of d_out
  float* out_node = (float*)d_out;
  float* e_cur = out_node + (size_t)NNODES * CH;

  const int NELEM = NNODES * CH;   // 2,560,000
  const int EELEM = NEDGES * CH;   // 40,960,000

  copyf<<<(NELEM + 255) / 256, 256, 0, stream>>>(x_cur, node_features, NELEM);
  copyf<<<(EELEM + 255) / 256, 256, 0, stream>>>(e_cur, edge_features, EELEM);

  // Pack all weight matrices to fragment-order bf16 (KT = K/32)
  unsigned* wpE1[2]; unsigned* wpE2[2]; unsigned* wpN1[2]; unsigned* wpN2[2];
  for (int c = 0; c < 2; c++) {
    int pb = 3 + 16 * c;
    wpE1[c] = walloc(8);  pack_w<<<8, 256, 0, stream>>>(F(pb + 0),  wpE1[c], 8);
    wpE2[c] = walloc(4);  pack_w<<<4, 256, 0, stream>>>(F(pb + 4),  wpE2[c], 4);
    wpN1[c] = walloc(12); pack_w<<<12, 256, 0, stream>>>(F(pb + 8), wpN1[c], 12);
    wpN2[c] = walloc(4);  pack_w<<<4, 256, 0, stream>>>(F(pb + 12), wpN2[c], 4);
  }
  unsigned* wpHEB = walloc(8); pack_w<<<8, 256, 0, stream>>>(F(35), wpHEB, 8);
  unsigned* wpHEO = walloc(4); pack_w<<<4, 256, 0, stream>>>(F(39), wpHEO, 4);
  unsigned* wpHNB = walloc(4); pack_w<<<4, 256, 0, stream>>>(F(41), wpHNB, 4);
  unsigned* wpHNO = walloc(4); pack_w<<<4, 256, 0, stream>>>(F(45), wpHNO, 4);

  for (int c = 0; c < 2; c++) {
    int pb = 3 + 16 * c;
    zerof<<<(NELEM + 255) / 256, 256, 0, stream>>>(agg, NELEM);
    conv_edge_kernel<<<NEDGES / 16, 32, 0, stream>>>(
        x_cur, e_cur, eidx,
        wpN1[c], F(pb + 9),  F(pb + 10), F(pb + 11),
        wpN2[c], F(pb + 13), F(pb + 14), F(pb + 15),
        wpE1[c], F(pb + 1),  F(pb + 2),  F(pb + 3),
        wpE2[c], F(pb + 5),  F(pb + 6),  F(pb + 7),
        agg);
    addf<<<(NELEM + 255) / 256, 256, 0, stream>>>(x_cur, agg, NELEM);
  }

  node_head_kernel<<<NNODES / 16, 32, 0, stream>>>(
      x_cur, wpHNB, F(42), F(43), F(44), wpHNO, F(46), out_node);

  edge_head_kernel<<<NEDGES / 16, 32, 0, stream>>>(
      x_cur, e_cur, eidx, wpHEB, F(36), F(37), F(38), wpHEO, F(40));
}
